// ConvLayer_7138235646006
// MI455X (gfx1250) — matrix operand order repaired and checked by reading, not compiled, not run on hardware
//
#include <hip/hip_runtime.h>
#include <hip/hip_bf16.h>
#include <math.h>

typedef __attribute__((ext_vector_type(16))) _Float16 v16h;
typedef __attribute__((ext_vector_type(8)))  float    v8f;
typedef __attribute__((ext_vector_type(4)))  uint32_t u32x4;
typedef __attribute__((ext_vector_type(8)))  uint32_t u32x8;

#define NNODES 4096
#define MUL0 32
#define MUL1 16
#define NB 20
#define FDIM 80            // MUL0 + 3*MUL1
#define STG 120            // staging floats per edge: h(20) xs(32) t(16) xv0(16) xv1(16) xv2(16) Y1(3) pad(1)
// staging offsets
#define OFF_H 0
#define OFF_XS 20
#define OFF_T 52
#define OFF_XV 68
#define OFF_Y 116

// B fragment store: frag = 512 halfs (32 lanes x 16 slots)
// frags 0..59  : B_s  (kstep 0..29, ntile 0..1)  -> out_s
// frags 60..79 : B_p  (k 0..19)                  -> p
// frags 80..89 : B_v  (kv 0..9)                  -> q_i
#define NFRAG 90
#define FRAG_HALFS 512

__device__ __forceinline__ v8f wmma16(v16h a, v16h b, v8f c) {
    return __builtin_amdgcn_wmma_f32_16x16x32_f16(false, a, false, b, (short)0, c, false, false);
}

// ---------------------------------------------------------------------------
// Repack mlp_w2 (20 x 2304, f32) into fragment-ready f16 B tiles (norm folded)
// ---------------------------------------------------------------------------
__global__ void prep_B(const float* __restrict__ w2, _Float16* __restrict__ bf) {
    int idx = blockIdx.x * blockDim.x + threadIdx.x;
    if (idx >= NFRAG * FRAG_HALFS) return;
    int frag = idx >> 9;
    int el   = idx & 511;
    int lane = el >> 4;
    int slot = el & 15;
    int n = lane & 15;
    int klocal = ((slot & 8) << 1) + (slot & 7) + ((lane >= 16) ? 8 : 0);   // B layout: lanes 0-15 K=0..7,16..23, lanes 16-31 K=8..15,24..31

    const float C    = 0.14433757f;   // 48^-0.5
    const float C110 = 0.08333333f;   // C / sqrt(3)
    const float I20  = 0.22360680f;   // 20^-0.5

    float val = 0.f;
    if (frag < 60) {                       // B_s : K-dim 960 = [G1(640) | G4(320)]
        int kstep = frag >> 1;
        int ntile = frag & 1;
        int w = ntile * 16 + n;
        int K = kstep * 32 + klocal;
        if (K < 640) {                     // W1 region: A = h[b]*xs[u], K = b*32+u
            int b = K >> 5, u = K & 31;
            val = C * I20 * w2[b * 2304 + u * 32 + w];
        } else {                           // W4 region: A = h[b]*t[u], K-640 = b*16+u
            int L = K - 640;
            int b = L >> 4, u = L & 15;
            val = C110 * I20 * w2[b * 2304 + 1792 + u * 32 + w];
        }
    } else if (frag < 80) {                // B_p : K = b*32+u, W2 region, n<16
        int K = (frag - 60) * 32 + klocal;
        int b = K >> 5, u = K & 31;
        val = C * I20 * w2[b * 2304 + 1024 + u * 16 + n];
    } else {                               // B_v : K = b*16+u, W3 region
        int K = (frag - 80) * 32 + klocal;
        int b = K >> 4, u = K & 15;
        val = C * I20 * w2[b * 2304 + 1536 + u * 16 + n];
    }
    bf[idx] = (_Float16)val;
}

// ---------------------------------------------------------------------------
// Per-edge geometry + radial MLP -> staging rows (f32), zero padded
// ---------------------------------------------------------------------------
__global__ void edge_prep(const float* __restrict__ pos, const float* __restrict__ f_in,
                          const float* __restrict__ w1, const int* __restrict__ esrc,
                          const int* __restrict__ edst, float* __restrict__ stage,
                          int E, int Epad) {
    int e = blockIdx.x * blockDim.x + threadIdx.x;
    if (e >= Epad) return;
    float* row = stage + (size_t)e * STG;
    if (e >= E) {
        #pragma unroll 4
        for (int i = 0; i < STG; ++i) row[i] = 0.f;
        return;
    }
    int s = esrc[e], d = edst[e];
    float e0 = pos[d * 3 + 0] - pos[s * 3 + 0];
    float e1 = pos[d * 3 + 1] - pos[s * 3 + 1];
    float e2 = pos[d * 3 + 2] - pos[s * 3 + 2];
    float r = sqrtf(e0 * e0 + e1 * e1 + e2 * e2);
    float inv = 1.f / r;
    const float SQ3 = 1.7320508f;
    float Y0 = SQ3 * e1 * inv;       // unit[[1,2,0]]
    float Y1 = SQ3 * e2 * inv;
    float Y2 = SQ3 * e0 * inv;

    const float EMB_SC = 1.14136f * 7.3890561f * 4.4721360f;  // SMOOTH_C * sqrt(20)
    float emb[NB];
    #pragma unroll
    for (int b = 0; b < NB; ++b) {
        float diff = r * 21.f - (float)(b + 1);
        float a1 = diff + 1.f, a2 = 1.f - diff;
        float s1 = (a1 > 0.f) ? expf(-1.f / a1) : 0.f;
        float s2 = (a2 > 0.f) ? expf(-1.f / a2) : 0.f;
        emb[b] = EMB_SC * s1 * s2;
    }
    const float I20 = 0.22360680f, SQ2 = 1.4142136f;
    #pragma unroll 4
    for (int j = 0; j < NB; ++j) {
        float acc = 0.f;
        #pragma unroll
        for (int b = 0; b < NB; ++b) acc += emb[b] * w1[b * NB + j];
        acc *= I20;
        row[OFF_H + j] = (acc > 0.f) ? SQ2 * acc : 0.f;
    }
    const float* x = f_in + (size_t)s * FDIM;
    #pragma unroll 8
    for (int u = 0; u < MUL0; ++u) row[OFF_XS + u] = x[u];
    #pragma unroll
    for (int u = 0; u < MUL1; ++u) {
        float v0 = x[MUL0 + u * 3 + 0];
        float v1 = x[MUL0 + u * 3 + 1];
        float v2 = x[MUL0 + u * 3 + 2];
        row[OFF_T + u]       = v0 * Y0 + v1 * Y1 + v2 * Y2;
        row[OFF_XV + u]      = v0;
        row[OFF_XV + 16 + u] = v1;
        row[OFF_XV + 32 + u] = v2;
    }
    row[OFF_Y + 0] = Y0; row[OFF_Y + 1] = Y1; row[OFF_Y + 2] = Y2;
    row[119] = 0.f;
}

// ---------------------------------------------------------------------------
// Main fused kernel: one wave = 16-edge tile. TDM stages the tile into LDS,
// outer-product A fragments feed WMMA against constant B fragments, atomic
// scatter accumulates into node features.
// ---------------------------------------------------------------------------
__global__ void __launch_bounds__(128) conv_main(const float* __restrict__ stage,
                                                 const _Float16* __restrict__ bfrag,
                                                 const int* __restrict__ edst,
                                                 float* __restrict__ f,
                                                 int E, int ntiles) {
    __shared__ float lds[4][16 * STG];
    int lane = threadIdx.x & 31;
    int wave = threadIdx.x >> 5;
    int tile = blockIdx.x * 4 + wave;
    bool active = (tile < ntiles);
    if (!active) tile = 0;                       // wave-uniform clamp; EXEC stays full
    float* st = lds[wave];

    // ---- Tensor Data Mover: DMA this wave's 16-edge staging block (7680 B)
    //      into its private LDS region (D# per CDNA5 ISA ch.8) ----
    {
        uint64_t ga = (uint64_t)(uintptr_t)(stage + (size_t)tile * 16 * STG);
        uint32_t lds_off = (uint32_t)(uintptr_t)st;      // LDS aperture: low 32 bits = LDS byte offset
        const uint32_t DW = 16 * STG;                    // 1920 dwords per tile
        u32x4 g0;
        g0[0] = __builtin_amdgcn_readfirstlane(1u);                                   // count=1, user mode
        g0[1] = __builtin_amdgcn_readfirstlane(lds_off);                              // lds_addr
        g0[2] = __builtin_amdgcn_readfirstlane((uint32_t)ga);                         // global_addr[31:0]
        g0[3] = __builtin_amdgcn_readfirstlane(
                    ((uint32_t)(ga >> 32) & 0x01FFFFFFu) | 0x80000000u);              // addr[56:32] | type=2
        u32x8 g1;
        g1[0] = __builtin_amdgcn_readfirstlane(2u << 16);            // data_size=4B, no multicast/pad
        g1[1] = __builtin_amdgcn_readfirstlane((DW & 0xFFFFu) << 16);// tensor_dim0[15:0]
        g1[2] = __builtin_amdgcn_readfirstlane((DW >> 16) | (1u << 16)); // dim0[31:16] | tensor_dim1[15:0]=1
        g1[3] = __builtin_amdgcn_readfirstlane(DW << 16);            // dim1[31:16]=0 | tile_dim0=1920
        g1[4] = __builtin_amdgcn_readfirstlane(1u);                  // tile_dim1=1, tile_dim2=0
        g1[5] = __builtin_amdgcn_readfirstlane(DW);                  // tensor_dim0_stride[31:0]
        g1[6] = __builtin_amdgcn_readfirstlane(0u);
        g1[7] = __builtin_amdgcn_readfirstlane(0u);
        asm volatile("tensor_load_to_lds %0, %1" :: "s"(g0), "s"(g1) : "memory");
        __builtin_amdgcn_s_wait_tensorcnt(0);
        __builtin_amdgcn_wave_barrier();
    }

    int r   = lane & 15;                         // edge row within tile (A-matrix M)
    int hiK = (lane & 16) >> 1;                  // +8 K shift for upper half-wave
    const float* hrow = st + r * STG;
    const v16h* B = (const v16h*)bfrag;          // frag f, lane l -> B[f*32 + l]

    v8f S0 = {}, S1 = {}, P = {};
    v8f Q[3] = {};

    // ---- loop A (k=0..19): G1 = h[k] (x) xs -> S0,S1,P. B loads issued first
    //      so the ~30-op fragment build hides their latency. ----
    for (int k = 0; k < 20; ++k) {
        v16h b0 = B[(k * 2 + 0) * 32 + lane];
        v16h b1 = B[(k * 2 + 1) * 32 + lane];
        v16h bp = B[(60 + k) * 32 + lane];
        float hv = hrow[OFF_H + k];
        v16h a;
        #pragma unroll
        for (int s = 0; s < 16; ++s) {
            int kl = ((s & 8) << 1) + (s & 7) + hiK;   // 16-bit A-layout K mapping
            a[s] = (_Float16)(hv * hrow[OFF_XS + kl]);
        }
        S0 = wmma16(a, b0, S0);
        S1 = wmma16(a, b1, S1);
        P  = wmma16(a, bp, P);
    }
    // ---- loop B (k=20..29): G4 = h (x) t -> S0,S1 ----
    for (int k = 0; k < 10; ++k) {
        v16h b0 = B[((k + 20) * 2 + 0) * 32 + lane];
        v16h b1 = B[((k + 20) * 2 + 1) * 32 + lane];
        int base = k * 32;
        v16h a;
        #pragma unroll
        for (int s = 0; s < 16; ++s) {
            int kl = ((s & 8) << 1) + (s & 7) + hiK;
            int K = base + kl;
            a[s] = (_Float16)(hrow[OFF_H + (K >> 4)] * hrow[OFF_T + (K & 15)]);
        }
        S0 = wmma16(a, b0, S0);
        S1 = wmma16(a, b1, S1);
    }
    // ---- loop C: G3i = h (x) xv_i -> Q[i] ----
    #pragma unroll
    for (int i = 0; i < 3; ++i) {
        const float* xv = hrow + OFF_XV + 16 * i;
        for (int kv = 0; kv < 10; ++kv) {
            v16h bv = B[(80 + kv) * 32 + lane];
            v16h a;
            #pragma unroll
            for (int s = 0; s < 16; ++s) {
                int kl = ((s & 8) << 1) + (s & 7) + hiK;
                int K = kv * 32 + kl;
                a[s] = (_Float16)(hrow[OFF_H + (K >> 4)] * xv[K & 15]);
            }
            Q[i] = wmma16(a, bv, Q[i]);
        }
    }

    // ---- scatter (C/D layout: lane = column N, rows M = v + 8*(lane>=16)) ----
    if (active) {
        int col = lane & 15;
        int hi  = lane >> 4;
        #pragma unroll
        for (int v = 0; v < 8; ++v) {
            int er = v + hi * 8;
            int e  = tile * 16 + er;
            int d  = (e < E) ? edst[e] : 0;      // padded rows contribute exact zeros
            float* fd = f + (size_t)d * FDIM;
            atomicAdd(fd + col,      S0[v]);
            atomicAdd(fd + 16 + col, S1[v]);
            float y0 = st[er * STG + OFF_Y + 0];
            float y1 = st[er * STG + OFF_Y + 1];
            float y2 = st[er * STG + OFF_Y + 2];
            float p = P[v];
            atomicAdd(fd + 32 + col * 3 + 0, p * y0 + Q[0][v]);
            atomicAdd(fd + 32 + col * 3 + 1, p * y1 + Q[1][v]);
            atomicAdd(fd + 32 + col * 3 + 2, p * y2 + Q[2][v]);
        }
    }
}

// ---------------------------------------------------------------------------
// Zero accumulator + BN stats + BN apply
// ---------------------------------------------------------------------------
__global__ void zero_f(float* __restrict__ p, int n) {
    int i = blockIdx.x * blockDim.x + threadIdx.x;
    if (i < n) p[i] = 0.f;
}

// stats: [0..31]=sum_s, [32..63]=sum_s^2, [64..79]=sum |v|^2
__global__ void bn_stats(const float* __restrict__ f, float* __restrict__ stats) {
    __shared__ float ls[80];
    int t = threadIdx.x;
    if (t < 80) ls[t] = 0.f;
    __syncthreads();
    int n = blockIdx.x * blockDim.x + t;
    if (n < NNODES) {
        const float* row = f + (size_t)n * FDIM;
        #pragma unroll 8
        for (int c = 0; c < MUL0; ++c) {
            float x = row[c];
            atomicAdd(&ls[c], x);
            atomicAdd(&ls[32 + c], x * x);
        }
        #pragma unroll
        for (int w = 0; w < MUL1; ++w) {
            float a = row[32 + w * 3 + 0];
            float b = row[32 + w * 3 + 1];
            float c2 = row[32 + w * 3 + 2];
            atomicAdd(&ls[64 + w], a * a + b * b + c2 * c2);
        }
    }
    __syncthreads();
    if (t < 80) atomicAdd(&stats[t], ls[t]);
}

__global__ void bn_apply(const float* __restrict__ f, const float* __restrict__ stats,
                         const float* __restrict__ gamma, const float* __restrict__ beta,
                         float* __restrict__ out) {
    int i = blockIdx.x * blockDim.x + threadIdx.x;
    if (i >= NNODES * FDIM) return;
    int c = i % FDIM;
    float x = f[i];
    const float eps = 1e-5f;
    const float invN = 1.f / (float)NNODES;
    if (c < MUL0) {
        float mean = stats[c] * invN;
        float var  = stats[32 + c] * invN - mean * mean;
        float scale = rsqrtf(var + eps) * gamma[c];
        out[i] = (x - mean) * scale + beta[c];
    } else {
        int w = (c - MUL0) / 3;
        float ms = stats[64 + w] * (invN / 3.f);
        out[i] = x * rsqrtf(ms + eps) * gamma[MUL0 + w];
    }
}

// ---------------------------------------------------------------------------
extern "C" void kernel_launch(void* const* d_in, const int* in_sizes, int n_in,
                              void* d_out, int out_size, void* d_ws, size_t ws_size,
                              hipStream_t stream) {
    const float* pos   = (const float*)d_in[0];
    const float* f_in  = (const float*)d_in[1];
    const float* w1    = (const float*)d_in[2];
    const float* w2    = (const float*)d_in[3];
    const float* gamma = (const float*)d_in[4];
    const float* beta  = (const float*)d_in[5];
    const int*   esrc  = (const int*)d_in[6];
    const int*   edst  = (const int*)d_in[7];
    int E    = in_sizes[6];
    int Epad = (E + 15) & ~15;
    int ntiles = Epad / 16;

    char* ws = (char*)d_ws;
    _Float16* bfrag = (_Float16*)ws;                                   // 92160 B
    float* f     = (float*)(ws + 92160);                               // N*80*4
    float* stats = (float*)(ws + 92160 + (size_t)NNODES * FDIM * 4);   // 512 B
    float* stage = (float*)(ws + 92160 + (size_t)NNODES * FDIM * 4 + 512);

    prep_B<<<(NFRAG * FRAG_HALFS + 255) / 256, 256, 0, stream>>>(w2, bfrag);
    zero_f<<<(NNODES * FDIM + 128 + 255) / 256, 256, 0, stream>>>(f, NNODES * FDIM + 128);
    if (Epad > 0) {
        edge_prep<<<(Epad + 127) / 128, 128, 0, stream>>>(pos, f_in, w1, esrc, edst, stage, E, Epad);
        conv_main<<<(ntiles + 3) / 4, 128, 0, stream>>>(stage, bfrag, edst, f, E, ntiles);
    }
    bn_stats<<<(NNODES + 255) / 256, 256, 0, stream>>>(f, stats);
    bn_apply<<<(NNODES * FDIM + 255) / 256, 256, 0, stream>>>(f, stats, gamma, beta, (float*)d_out);
}
